// WindowAttention_72825465471153
// MI455X (gfx1250) — matrix operand order repaired and checked by reading, not compiled, not run on hardware
//
#include <hip/hip_runtime.h>

typedef _Float16 v16h __attribute__((ext_vector_type(16)));
typedef _Float16 v8h  __attribute__((ext_vector_type(8)));
typedef float    v8f  __attribute__((ext_vector_type(8)));

#define B_    8
#define CH    512
#define NHEAD 8
#define HD    64
#define IMG   128
#define WSZ   8
#define TOK   64
#define NWIN  16

// A-operand (16x32 f16): lane holds row (row_base + lane%16); K-interleave per ISA:
// VGPR0-3 = K in [kb + 8*half, +8), VGPR4-7 = K in [kb + 16 + 8*half, +8)
__device__ __forceinline__ v16h frag_a(const _Float16* X, int row_base, int kb,
                                       int n, int half) {
  const _Float16* p = X + (row_base + n) * 64 + kb + 8 * half;
  union { v16h v; v8h h[2]; } u;
  u.h[0] = *(const v8h*)p;
  u.h[1] = *(const v8h*)(p + 16);
  return u.v;
}

// B-operand (32x16 f16): lane holds column (col_base + lane%16); K = [kb + 8*half, +8) then [kb + 16 + 8*half, +8)
// X laid out [column][Kdim] with row length 64 -> two 16-byte runs per lane.
__device__ __forceinline__ v16h frag_b(const _Float16* X, int col_base, int kb,
                                       int n, int half) {
  const _Float16* p = X + (col_base + n) * 64 + kb + 8 * half;
  union { v16h v; v8h h[2]; } u;
  u.h[0] = *(const v8h*)p;
  u.h[1] = *(const v8h*)(p + 16);
  return u.v;
}

__global__ __launch_bounds__(128)
void win_attn_kernel(const float* __restrict__ qg, const float* __restrict__ kg,
                     const float* __restrict__ vg, float* __restrict__ outg) {
  __shared__ __align__(16) _Float16 sQ[TOK * HD];   // token-major, pre-scaled
  __shared__ __align__(16) _Float16 sK[TOK * HD];   // token-major (N=token, K=d)
  __shared__ __align__(16) _Float16 sV[HD * TOK];   // d-major    (N=d,     K=token)
  __shared__ __align__(16) _Float16 sP[TOK * TOK];  // token-major (q rows)
  __shared__ __align__(16) float    sO[TOK * HD];   // token-major f32

  const int tid  = threadIdx.x;
  const int wave = tid >> 5;
  const int lane = tid & 31;
  const int n    = lane & 15;
  const int half = lane >> 4;

  const int wid  = blockIdx.x;          // 0 .. 16383
  const int ww   = wid & 15;
  const int wh   = (wid >> 4) & 15;
  const int head = (wid >> 8) & 7;
  const int b    = wid >> 11;
  const int c0   = head * HD;

  const size_t base = (((size_t)(b * CH + c0)) * IMG + wh * WSZ) * IMG + ww * WSZ;
  const float scale = 0.125f;           // 64^-0.5

  // ---- cooperative staging: HBM -> LDS (f32 -> f16), coalesced along W ----
  #pragma unroll 4
  for (int i = 0; i < 32; ++i) {
    const int e = i * 128 + tid;
    const int d = e >> 6;               // channel within head
    const int t = e & 63;               // token within window
    const size_t g = base + (size_t)d * (IMG * IMG) + (t >> 3) * IMG + (t & 7);
    sQ[t * HD + d]  = (_Float16)(qg[g] * scale);
    sK[t * HD + d]  = (_Float16)kg[g];
    sV[d * TOK + t] = (_Float16)vg[g];
  }
  __syncthreads();

  const int tr = wave * 16;             // this wave's q-token slab

  // ---- S = Q * K^T  (4 column tiles x 2 k-steps) ----
  v8f s[4] = {};
  #pragma unroll
  for (int ks = 0; ks < 2; ++ks) {
    const int kb = ks * 32;
    const v16h a = frag_a(sQ, tr, kb, n, half);
    #pragma unroll
    for (int ct = 0; ct < 4; ++ct) {
      const v16h bb = frag_b(sK, ct * 16, kb, n, half);
      s[ct] = __builtin_amdgcn_wmma_f32_16x16x32_f16(
          false, a, false, bb, (short)0, s[ct], false, false);
    }
  }

  // ---- row softmax: C-layout row r lives in VGPR r across the 16 lanes of a half ----
  #pragma unroll
  for (int r = 0; r < 8; ++r) {
    float mx = fmaxf(fmaxf(s[0][r], s[1][r]), fmaxf(s[2][r], s[3][r]));
    #pragma unroll
    for (int off = 8; off > 0; off >>= 1)
      mx = fmaxf(mx, __shfl_xor(mx, off, 32));
    const float e0 = __expf(s[0][r] - mx);
    const float e1 = __expf(s[1][r] - mx);
    const float e2 = __expf(s[2][r] - mx);
    const float e3 = __expf(s[3][r] - mx);
    float sum = (e0 + e1) + (e2 + e3);
    #pragma unroll
    for (int off = 8; off > 0; off >>= 1)
      sum += __shfl_xor(sum, off, 32);
    const float inv = __builtin_amdgcn_rcpf(sum);
    const int prow = (tr + 8 * half + r) * TOK;
    sP[prow +  0 + n] = (_Float16)(e0 * inv);
    sP[prow + 16 + n] = (_Float16)(e1 * inv);
    sP[prow + 32 + n] = (_Float16)(e2 * inv);
    sP[prow + 48 + n] = (_Float16)(e3 * inv);
  }
  __syncthreads();

  // ---- O = P * V  (4 d tiles x 2 k-steps over the 64 k-tokens) ----
  v8f o[4] = {};
  #pragma unroll
  for (int ks = 0; ks < 2; ++ks) {
    const int kb = ks * 32;
    const v16h a = frag_a(sP, tr, kb, n, half);
    #pragma unroll
    for (int dt = 0; dt < 4; ++dt) {
      const v16h bb = frag_b(sV, dt * 16, kb, n, half);
      o[dt] = __builtin_amdgcn_wmma_f32_16x16x32_f16(
          false, a, false, bb, (short)0, o[dt], false, false);
    }
  }

  // ---- O (C-layout) -> LDS token-major ----
  #pragma unroll
  for (int dt = 0; dt < 4; ++dt)
    #pragma unroll
    for (int r = 0; r < 8; ++r)
      sO[(tr + 8 * half + r) * HD + dt * 16 + n] = o[dt][r];
  __syncthreads();

  // ---- cooperative store: LDS -> HBM, coalesced along W ----
  #pragma unroll 4
  for (int i = 0; i < 32; ++i) {
    const int e = i * 128 + tid;
    const int d = e >> 6;
    const int t = e & 63;
    const size_t g = base + (size_t)d * (IMG * IMG) + (t >> 3) * IMG + (t & 7);
    outg[g] = sO[t * HD + d];
  }
}

extern "C" void kernel_launch(void* const* d_in, const int* in_sizes, int n_in,
                              void* d_out, int out_size, void* d_ws, size_t ws_size,
                              hipStream_t stream) {
  const float* q = (const float*)d_in[0];
  const float* k = (const float*)d_in[1];
  const float* v = (const float*)d_in[2];
  float* out = (float*)d_out;
  const int nblocks = B_ * NHEAD * NWIN * NWIN;   // 16384 windows
  win_attn_kernel<<<dim3(nblocks), dim3(128), 0, stream>>>(q, k, v, out);
}
